// MiniGridPPOMixtureMHNetwork_68719477113
// MI455X (gfx1250) — matrix operand order repaired and checked by reading, not compiled, not run on hardware
//
#include <hip/hip_runtime.h>
#include <hip/hip_bf16.h>

#define Bsz 4096
#define Ee  4
#define Cc  10
#define Dd  1024
#define NFf 128
#define NOo 7
#define EPSf 1e-8f

typedef __attribute__((ext_vector_type(16))) _Float16 v16h;
typedef __attribute__((ext_vector_type(8)))  _Float16 half8;
typedef __attribute__((ext_vector_type(8)))  float    v8f;

// ---------------------------------------------------------------- prep
__global__ void __launch_bounds__(256) prep_kernel(const float* __restrict__ h1_w,
                                                   _Float16* __restrict__ h1w_h,
                                                   int* __restrict__ counts, int n) {
  int i = blockIdx.x * blockDim.x + threadIdx.x;
  if (i < n) h1w_h[i] = (_Float16)h1_w[i];
  if (blockIdx.x == 0 && threadIdx.x < Cc) counts[threadIdx.x] = 0;
}

// ---------------------------------------------------------------- classify
__global__ void __launch_bounds__(256) classify_kernel(const int* __restrict__ cvec,
                                                       int* __restrict__ counts,
                                                       int* __restrict__ idxlist) {
  int b = blockIdx.x * blockDim.x + threadIdx.x;
  if (b < Bsz) {
    int cls  = cvec[b];
    int slot = atomicAdd(&counts[cls], 1);
    idxlist[cls * Bsz + slot] = b;
  }
}

// ---------------------------------------------------------------- CNN + GS + mix
__device__ __forceinline__ float blockDot(const float* a, const float* b,
                                          float* red, int tid) {
  float s = 0.f;
  for (int d = tid; d < Dd; d += 256) s += a[d] * b[d];
  // wave32 shuffle reduction
  for (int off = 16; off > 0; off >>= 1) s += __shfl_down(s, off, 32);
  if ((tid & 31) == 0) red[tid >> 5] = s;
  __syncthreads();
  if (tid == 0) {
    float t = 0.f;
    #pragma unroll
    for (int i = 0; i < 8; ++i) t += red[i];
    red[0] = t;
  }
  __syncthreads();
  float r = red[0];
  __syncthreads();
  return r;
}

__global__ void __launch_bounds__(256) cnn_gs_mix_kernel(
    const float* __restrict__ state, const int* __restrict__ cvec,
    const float* __restrict__ te_w,
    const float* __restrict__ w1, const float* __restrict__ b1,
    const float* __restrict__ w2, const float* __restrict__ b2,
    const float* __restrict__ w3, const float* __restrict__ b3,
    _Float16* __restrict__ mix_h) {
  __shared__ float sState[147];
  __shared__ float sC1[Ee][576];   // 16 x 6 x 6
  __shared__ float sC2[Ee][800];   // 32 x 5 x 5
  __shared__ float sF[Ee][Dd];     // 64 x 4 x 4 flattened
  __shared__ float sRed[8];

  const int b   = blockIdx.x;
  const int tid = threadIdx.x;
  const int e   = tid >> 6;   // expert 0..3 (64 threads each)
  const int lt  = tid & 63;

  for (int i = tid; i < 147; i += 256) sState[i] = state[(size_t)b * 147 + i];
  __syncthreads();

  // conv1: (3,7,7) -> (16,6,6), relu
  for (int idx = lt; idx < 576; idx += 64) {
    int o = idx / 36, rem = idx % 36, y = rem / 6, x = rem % 6;
    float acc = b1[e * 16 + o];
    const float* wp = w1 + (size_t)((e * 16 + o) * 3) * 4;
    #pragma unroll
    for (int i = 0; i < 3; ++i)
      #pragma unroll
      for (int ky = 0; ky < 2; ++ky)
        #pragma unroll
        for (int kx = 0; kx < 2; ++kx)
          acc += wp[i * 4 + ky * 2 + kx] * sState[i * 49 + (y + ky) * 7 + (x + kx)];
    sC1[e][idx] = fmaxf(acc, 0.f);
  }
  __syncthreads();

  // conv2: (16,6,6) -> (32,5,5), relu
  for (int idx = lt; idx < 800; idx += 64) {
    int o = idx / 25, rem = idx % 25, y = rem / 5, x = rem % 5;
    float acc = b2[e * 32 + o];
    const float* wp = w2 + (size_t)((e * 32 + o) * 16) * 4;
    for (int i = 0; i < 16; ++i)
      #pragma unroll
      for (int ky = 0; ky < 2; ++ky)
        #pragma unroll
        for (int kx = 0; kx < 2; ++kx)
          acc += wp[i * 4 + ky * 2 + kx] * sC1[e][i * 36 + (y + ky) * 6 + (x + kx)];
    sC2[e][idx] = fmaxf(acc, 0.f);
  }
  __syncthreads();

  // conv3: (32,5,5) -> (64,4,4), no relu; flatten d = o*16 + y*4 + x
  for (int idx = lt; idx < Dd; idx += 64) {
    int o = idx >> 4, rem = idx & 15, y = rem >> 2, x = rem & 3;
    float acc = b3[e * 64 + o];
    const float* wp = w3 + (size_t)((e * 64 + o) * 32) * 4;
    for (int i = 0; i < 32; ++i)
      #pragma unroll
      for (int ky = 0; ky < 2; ++ky)
        #pragma unroll
        for (int kx = 0; kx < 2; ++kx)
          acc += wp[i * 4 + ky * 2 + kx] * sC2[e][i * 25 + (y + ky) * 5 + (x + kx)];
    sF[e][idx] = acc;
  }
  __syncthreads();

  // classical Gram-Schmidt (projections taken against the un-modified vector,
  // matching the reference's einsum ordering)
  {
    float n0 = blockDot(sF[0], sF[0], sRed, tid);
    float inv = 1.f / (sqrtf(n0) + EPSf);
    for (int d = tid; d < Dd; d += 256) sF[0][d] *= inv;
    __syncthreads();
  }
  for (int i = 1; i < Ee; ++i) {
    float pj[3];
    for (int j = 0; j < i; ++j) pj[j] = blockDot(sF[i], sF[j], sRed, tid);
    for (int d = tid; d < Dd; d += 256) {
      float v = sF[i][d];
      for (int j = 0; j < i; ++j) v -= pj[j] * sF[j][d];
      sF[i][d] = v;
    }
    __syncthreads();
    float nn = blockDot(sF[i], sF[i], sRed, tid);
    float inv = 1.f / (sqrtf(nn) + EPSf);
    for (int d = tid; d < Dd; d += 256) sF[i][d] *= inv;
    __syncthreads();
  }

  // mix = tanh(sum_e te_w[e][c[b]] * basis_e), stored fp16 for the WMMA GEMM
  const int cls = cvec[b];
  const float g0 = te_w[0 * Cc + cls], g1 = te_w[1 * Cc + cls];
  const float g2 = te_w[2 * Cc + cls], g3 = te_w[3 * Cc + cls];
  for (int d = tid; d < Dd; d += 256) {
    float m = tanhf(g0 * sF[0][d] + g1 * sF[1][d] + g2 * sF[2][d] + g3 * sF[3][d]);
    mix_h[(size_t)b * Dd + d] = (_Float16)m;
  }
}

// ---------------------------------------------------------------- class-grouped WMMA GEMM + head
// Block = (class, 16-row tile). 8 waves; wave w owns features [16w, 16w+16).
// h1[b][f] = tanh(sum_d mix[b][d] * h1_w[cls][f][d] + h1_b[cls][f])
// out[b][o] = sum_f h1[b][f] * h2_w[cls][o][f] + h2_b[cls][o]
__global__ void __launch_bounds__(256) gemm_head_kernel(
    const _Float16* __restrict__ mix_h, const _Float16* __restrict__ h1w_h,
    const float* __restrict__ h1_b, const float* __restrict__ h2_w,
    const float* __restrict__ h2_b, const int* __restrict__ counts,
    const int* __restrict__ idxlist, float* __restrict__ out) {
  __shared__ __align__(16) _Float16 sA[16 * Dd];   // 32 KB gathered mix tile
  __shared__ __align__(16) float    sH[16 * NFf];  // 8 KB h1 tile
  __shared__ int sRow[16];
  __shared__ int sValid[16];

  const int cls      = blockIdx.y;
  const int count    = counts[cls];
  const int rowStart = blockIdx.x * 16;
  if (rowStart >= count) return;   // uniform per block

  const int tid = threadIdx.x;
  if (tid < 16) {
    int g = rowStart + tid;
    int valid = (g < count) ? 1 : 0;
    sRow[tid]   = valid ? idxlist[cls * Bsz + g] : idxlist[cls * Bsz]; // count>=1
    sValid[tid] = valid;
  }
  __syncthreads();

  // Stage A (16 rows x 1024 halves): 16 threads/row, 64 halves (8x16B) each
  {
    int r = tid >> 4, chunk = tid & 15;
    const uint4* src = (const uint4*)(mix_h + (size_t)sRow[r] * Dd + chunk * 64);
    uint4* dst = (uint4*)(sA + r * Dd + chunk * 64);
    #pragma unroll
    for (int i = 0; i < 8; ++i) dst[i] = src[i];
  }
  __syncthreads();

  const int lane  = tid & 31;
  const int wave  = tid >> 5;       // 0..7 -> feature strip
  const int lhalf = lane >> 4;      // 0/1
  const int lrow  = lane & 15;      // A row (M) / B column (N)

  const _Float16* brow = h1w_h + ((size_t)cls * NFf + wave * 16 + lrow) * Dd;

  v8f acc = {};
  for (int k0 = 0; k0 < Dd; k0 += 32) {
    __builtin_prefetch(brow + k0 + 128, 0, 1);   // stream next weight chunk
    // A fragment: lane holds row M=lrow, K = lhalf*16..+15 (contiguous, matching B)
    half8 lo = *(const half8*)(sA + lrow * Dd + k0 + lhalf * 16);
    half8 hi = *(const half8*)(sA + lrow * Dd + k0 + lhalf * 16 + 8);
    v16h a;
    #pragma unroll
    for (int i = 0; i < 8; ++i) { a[i] = lo[i]; a[i + 8] = hi[i]; }
    // B fragment: lane holds 16 consecutive K for column N=lrow -> contiguous 32B of the weight row
    v16h bf = *(const v16h*)(brow + k0 + lhalf * 16);
    acc = __builtin_amdgcn_wmma_f32_16x16x32_f16(false, a, false, bf,
                                                 (short)0, acc, false, false);
  }

  // bias + tanh; C/D layout: element i -> M = i + 8*lhalf, N = lrow
  const float bias = h1_b[cls * NFf + wave * 16 + lrow];
  #pragma unroll
  for (int i = 0; i < 8; ++i) {
    int m = i + 8 * lhalf;
    sH[m * NFf + wave * 16 + lrow] = tanhf(acc[i] + bias);
  }
  __syncthreads();

  // fused 7-wide head: 112 threads = 16 rows x 7 outputs
  if (tid < 16 * NOo) {
    int r = tid & 15, o = tid >> 4;
    if (sValid[r]) {
      const float* w = h2_w + ((size_t)cls * NOo + o) * NFf;
      float s = 0.f;
      #pragma unroll 4
      for (int f = 0; f < NFf; ++f) s += sH[r * NFf + f] * w[f];
      out[(size_t)sRow[r] * NOo + o] = s + h2_b[cls * NOo + o];
    }
  }
}

// ---------------------------------------------------------------- launch
extern "C" void kernel_launch(void* const* d_in, const int* in_sizes, int n_in,
                              void* d_out, int out_size, void* d_ws, size_t ws_size,
                              hipStream_t stream) {
  const float* state   = (const float*)d_in[0];
  const int*   cvec    = (const int*)d_in[1];
  const float* te_w    = (const float*)d_in[2];
  const float* conv1_w = (const float*)d_in[3];
  const float* conv1_b = (const float*)d_in[4];
  const float* conv2_w = (const float*)d_in[5];
  const float* conv2_b = (const float*)d_in[6];
  const float* conv3_w = (const float*)d_in[7];
  const float* conv3_b = (const float*)d_in[8];
  const float* h1_w    = (const float*)d_in[9];
  const float* h1_b    = (const float*)d_in[10];
  const float* h2_w    = (const float*)d_in[11];
  const float* h2_b    = (const float*)d_in[12];
  float* out = (float*)d_out;

  // workspace layout (~10.7 MB total)
  char* ws = (char*)d_ws;
  const size_t MIX_BYTES = (size_t)Bsz * Dd * sizeof(_Float16);        // 8 MB
  const size_t H1W_BYTES = (size_t)Cc * NFf * Dd * sizeof(_Float16);   // 2.5 MB
  _Float16* mix_h   = (_Float16*)ws;
  _Float16* h1w_h   = (_Float16*)(ws + MIX_BYTES);
  int*      counts  = (int*)(ws + MIX_BYTES + H1W_BYTES);
  int*      idxlist = counts + 64;  // 256B pad, then C*B ints

  const int n_h1w = Cc * NFf * Dd;
  prep_kernel<<<(n_h1w + 255) / 256, 256, 0, stream>>>(h1_w, h1w_h, counts, n_h1w);
  classify_kernel<<<(Bsz + 255) / 256, 256, 0, stream>>>(cvec, counts, idxlist);
  cnn_gs_mix_kernel<<<Bsz, 256, 0, stream>>>(state, cvec, te_w,
                                             conv1_w, conv1_b, conv2_w, conv2_b,
                                             conv3_w, conv3_b, mix_h);
  gemm_head_kernel<<<dim3(Bsz / 16, Cc), 256, 0, stream>>>(mix_h, h1w_h, h1_b,
                                                           h2_w, h2_b, counts,
                                                           idxlist, out);
}
